// ColBERTLoss_17282948399256
// MI455X (gfx1250) — matrix operand order repaired and checked by reading, not compiled, not run on hardware
//
#include <hip/hip_runtime.h>
#include <hip/hip_bf16.h>
#include <math.h>

// Problem constants (from reference setup_inputs)
#define BB    64      // batch
#define LQ    32      // query tokens
#define LDOC  256     // doc tokens
#define DIM   128     // embedding dim
#define CDOCS 128     // 2*B docs
#define TEMP_INV 20.0f   // 1/0.05
#define NEG_INF_F (-1.0e9f)

typedef __attribute__((ext_vector_type(16))) __bf16 v16bf;
typedef __attribute__((ext_vector_type(8)))  float  v8f;

union Frag32B {
    uint4  u[2];   // two 16-byte chunks
    v16bf  v;      // 16 bf16 = 32 bytes
};

// round-to-nearest-even float -> bf16 bits
static __device__ __forceinline__ unsigned short f2bf(float x) {
    unsigned u = __float_as_uint(x);
    u += 0x7FFFu + ((u >> 16) & 1u);
    return (unsigned short)(u >> 16);
}

// ---------------------------------------------------------------------------
// 1) L2-normalize tokens (D=128) and pack to bf16. One wave per token.
// ---------------------------------------------------------------------------
__global__ __launch_bounds__(256)
void normalize_bf16_kernel(const float* __restrict__ src,
                           unsigned short* __restrict__ dst, int ntok) {
    const int token = blockIdx.x * 8 + (threadIdx.x >> 5);
    const int lane  = threadIdx.x & 31;
    if (token >= ntok) return;
    const float4 x = *(const float4*)(src + (size_t)token * DIM + lane * 4);
    float ss = x.x * x.x + x.y * x.y + x.z * x.z + x.w * x.w;
    #pragma unroll
    for (int m = 16; m >= 1; m >>= 1) ss += __shfl_xor(ss, m, 32);
    const float scale = 1.0f / fmaxf(sqrtf(ss), 1e-12f);
    ushort4 o;
    o.x = f2bf(x.x * scale);
    o.y = f2bf(x.y * scale);
    o.z = f2bf(x.z * scale);
    o.w = f2bf(x.w * scale);
    *(ushort4*)(dst + (size_t)token * DIM + lane * 4) = o;
}

// ---------------------------------------------------------------------------
// 2) Fused MaxSim GEMM: per block (b,c) compute 32x256 token-sim tile with
//    v_wmma_f32_16x16x32_bf16, fuse max-over-doc-tokens + mean-over-query.
//    256 threads = 8 waves; wave w owns N-tiles {2w, 2w+1}, both M-tiles.
// ---------------------------------------------------------------------------
__global__ __launch_bounds__(256)
void maxsim_kernel(const unsigned short* __restrict__ qn,
                   const unsigned short* __restrict__ dn,
                   float* __restrict__ scores) {
    const int c    = blockIdx.x;            // doc 0..127
    const int b    = blockIdx.y;            // query row 0..63
    const int tid  = threadIdx.x;
    const int lane = tid & 31;
    const int w    = tid >> 5;              // wave 0..7
    const int lhalf = lane >> 4;            // 0 | 1
    const int l15   = lane & 15;

    __shared__ float partmax[8][LQ];

    const unsigned short* qbase = qn + (size_t)b * LQ * DIM;
    const unsigned short* dbase = dn + (size_t)c * LDOC * DIM;

    // A fragments: ISA 16-bit A 16x32 layout.
    // lane<16 holds rows M=l15, K in [0..8)U[16..24); lane>=16 shifted by 8.
    Frag32B A[2][4];
    #pragma unroll
    for (int mt = 0; mt < 2; ++mt) {
        const unsigned short* qr = qbase + (mt * 16 + l15) * DIM;
        #pragma unroll
        for (int ks = 0; ks < 4; ++ks) {
            const int k0 = ks * 32 + lhalf * 8;
            A[mt][ks].u[0] = *(const uint4*)(qr + k0);
            A[mt][ks].u[1] = *(const uint4*)(qr + k0 + 16);
        }
    }

    float rmax[2][8];
    #pragma unroll
    for (int mt = 0; mt < 2; ++mt)
        #pragma unroll
        for (int r = 0; r < 8; ++r) rmax[mt][r] = -3.0e38f;

    #pragma unroll
    for (int nti = 0; nti < 2; ++nti) {
        const int nt  = 2 * w + nti;
        // B fragment layout: lane holds doc column N=l15, K in [0..8)U[16..24); lane>=16 shifted by 8.
        const unsigned short* dr = dbase + (nt * 16 + l15) * DIM;
        v8f Cacc[2] = {};
        #pragma unroll
        for (int ks = 0; ks < 4; ++ks) {
            Frag32B Bf;
            const int k0 = ks * 32 + lhalf * 8;
            Bf.u[0] = *(const uint4*)(dr + k0);
            Bf.u[1] = *(const uint4*)(dr + k0 + 16);
            #pragma unroll
            for (int mt = 0; mt < 2; ++mt)
                Cacc[mt] = __builtin_amdgcn_wmma_f32_16x16x32_bf16(
                    false, A[mt][ks].v, false, Bf.v,
                    (short)0, Cacc[mt], false, false);
        }
        // C layout: VGPR r, lane l -> row = mt*16 + r + 8*(l>>4), col = l&15.
        // Max over cols (doc tokens) = butterfly max within each 16-lane half.
        #pragma unroll
        for (int mt = 0; mt < 2; ++mt)
            #pragma unroll
            for (int r = 0; r < 8; ++r) {
                float v = Cacc[mt][r];
                v = fmaxf(v, __shfl_xor(v, 1, 32));
                v = fmaxf(v, __shfl_xor(v, 2, 32));
                v = fmaxf(v, __shfl_xor(v, 4, 32));
                v = fmaxf(v, __shfl_xor(v, 8, 32));
                rmax[mt][r] = fmaxf(rmax[mt][r], v);
            }
    }

    // lanes 0 and 16 hold half-wave maxima for rows (mt*16+r) and (mt*16+r+8)
    if (l15 == 0) {
        #pragma unroll
        for (int mt = 0; mt < 2; ++mt)
            #pragma unroll
            for (int r = 0; r < 8; ++r)
                partmax[w][mt * 16 + r + 8 * lhalf] = rmax[mt][r];
    }
    __syncthreads();

    if (tid < LQ) {
        float v = partmax[0][tid];
        #pragma unroll
        for (int ww = 1; ww < 8; ++ww) v = fmaxf(v, partmax[ww][tid]);
        // mean over the 32 query rows
        #pragma unroll
        for (int m = 16; m >= 1; m >>= 1) v += __shfl_xor(v, m, 32);
        if (tid == 0)
            scores[b * CDOCS + c] = (v * (1.0f / LQ)) * TEMP_INV;
    }
}

// ---------------------------------------------------------------------------
// 3) Per-row masking + log-softmax cross-entropy. One 128-thread block per b.
// ---------------------------------------------------------------------------
__global__ __launch_bounds__(128)
void ce_kernel(const float* __restrict__ scores,
               const int* __restrict__ labels,
               float* __restrict__ losses) {
    const int b    = blockIdx.x;
    const int tid  = threadIdx.x;      // 0..127 = doc index
    const int lane = tid & 31;
    const int w    = tid >> 5;
    __shared__ float red[4];
    __shared__ float starget;

    float s = scores[b * CDOCS + tid];
    if (tid < BB && tid != b && labels[tid] == labels[b]) s = NEG_INF_F;
    if (tid == b) starget = s;

    // global max
    float m = s;
    #pragma unroll
    for (int off = 16; off >= 1; off >>= 1) m = fmaxf(m, __shfl_xor(m, off, 32));
    if (lane == 0) red[w] = m;
    __syncthreads();
    m = fmaxf(fmaxf(red[0], red[1]), fmaxf(red[2], red[3]));
    __syncthreads();

    // sum of exp
    float e = __expf(s - m);
    #pragma unroll
    for (int off = 16; off >= 1; off >>= 1) e += __shfl_xor(e, off, 32);
    if (lane == 0) red[w] = e;
    __syncthreads();

    if (tid == 0) {
        const float lse = logf(red[0] + red[1] + red[2] + red[3]) + m;
        losses[b] = lse - starget;     // = -log p[b,b]
    }
}

// ---------------------------------------------------------------------------
// 4) Mean of per-row losses -> scalar output.
// ---------------------------------------------------------------------------
__global__ void finalize_kernel(const float* __restrict__ losses,
                                float* __restrict__ out) {
    if (threadIdx.x == 0) {
        float s = 0.0f;
        for (int i = 0; i < BB; ++i) s += losses[i];
        out[0] = s * (1.0f / BB);
    }
}

extern "C" void kernel_launch(void* const* d_in, const int* in_sizes, int n_in,
                              void* d_out, int out_size, void* d_ws, size_t ws_size,
                              hipStream_t stream) {
    const float* q_emb = (const float*)d_in[0];   // [64,32,128]
    const float* p_emb = (const float*)d_in[1];   // [64,256,128]
    const float* n_emb = (const float*)d_in[2];   // [64,256,128]
    const int*   labels = (const int*)d_in[3];    // [64]

    char* ws = (char*)d_ws;
    const size_t qn_bytes = (size_t)BB * LQ * DIM * sizeof(unsigned short);      // 512 KB
    const size_t dn_bytes = (size_t)CDOCS * LDOC * DIM * sizeof(unsigned short); // 8 MB
    unsigned short* qn = (unsigned short*)ws;
    unsigned short* dn = (unsigned short*)(ws + qn_bytes);
    float* scores = (float*)(ws + qn_bytes + dn_bytes);            // 64*128 f32
    float* losses = scores + BB * CDOCS;                           // 64 f32

    const int q_tok = BB * LQ;          // 2048
    const int d_tok = BB * LDOC;        // 16384 per half

    normalize_bf16_kernel<<<q_tok / 8, 256, 0, stream>>>(q_emb, qn, q_tok);
    normalize_bf16_kernel<<<d_tok / 8, 256, 0, stream>>>(p_emb, dn, d_tok);
    normalize_bf16_kernel<<<d_tok / 8, 256, 0, stream>>>(
        n_emb, dn + (size_t)d_tok * DIM, d_tok);

    maxsim_kernel<<<dim3(CDOCS, BB), 256, 0, stream>>>(qn, dn, scores);
    ce_kernel<<<BB, 128, 0, stream>>>(scores, labels, losses);
    finalize_kernel<<<1, 64, 0, stream>>>(losses, (float*)d_out);
}
